// GraphAttention_15719580303494
// MI455X (gfx1250) — matrix operand order repaired and checked by reading, not compiled, not run on hardware
//
#include <hip/hip_runtime.h>

typedef __attribute__((ext_vector_type(16))) __bf16 v16bf;
typedef __attribute__((ext_vector_type(8)))  float  v8f;
typedef __attribute__((ext_vector_type(4)))  float  f4;

#define BT_     1600      // B*T
#define N_      5000
#define F_      128
#define H_      4
#define KP      5024      // K (=N) padded to multiple of 32
#define NP      5008      // output cols padded to multiple of 16
#define ATROWS  5120      // A^T rows padded to multiple of 512 (block col tile)
#define XTROWS  512
#define NEG_INF_F (-1e10f)
#define ALPHA   0.2f

// ---- float -> bf16 (round to nearest even) as raw u16 ----
__device__ __forceinline__ unsigned short f2bf(float f) {
    unsigned u = __float_as_uint(f);
    u += 0x7FFFu + ((u >> 16) & 1u);
    return (unsigned short)(u >> 16);
}

// ---- cast + right-pad columns with zeros: out[r, c] = (c<cin) ? in[r,c] : 0 ----
__global__ void k_cast_pad(const float* __restrict__ in, unsigned short* __restrict__ out,
                           int rows, int cin, int cout) {
    long long idx = (long long)blockIdx.x * blockDim.x + threadIdx.x;
    long long tot = (long long)rows * cout;
    if (idx >= tot) return;
    int r = (int)(idx / cout), c = (int)(idx % cout);
    float v = (c < cin) ? in[(long long)r * cin + c] : 0.0f;
    out[idx] = f2bf(v);
}

// ---- transpose + cast + pad: out[j, k] = (k<inRows && j<inCols) ? in[k, j] : 0 ----
__global__ void k_transpose_cast(const float* __restrict__ in, unsigned short* __restrict__ out,
                                 int outRows, int inRows, int inCols) {
    long long idx = (long long)blockIdx.x * blockDim.x + threadIdx.x;
    long long tot = (long long)outRows * KP;
    if (idx >= tot) return;
    int j = (int)(idx / KP), k = (int)(idx % KP);
    float v = (k < inRows && j < inCols) ? in[(long long)k * inCols + j] : 0.0f;
    out[idx] = f2bf(v);
}

union Frag { v16bf v; f4 q[2]; };

// C[M=BT_, :] = Abf[BT_, KP](bf16 row-major) x Btbf^T  (Btbf is B^T, row-major [cols, KP])
// Block: 256 thr = 8 waves; block owns a 32-row M-tile; wave w owns a 32x64 strip at
// colBase = blockIdx.y*512 + w*64. Two A-frags (rows rowTile / rowTile+16) share every B-frag.
__global__ __launch_bounds__(256)
void k_gemm_bf16(const unsigned short* __restrict__ Abf,
                 const unsigned short* __restrict__ Btbf,
                 float* __restrict__ C, int ldc, int nstore) {
    const int lane = threadIdx.x & 31;
    const int wave = threadIdx.x >> 5;
    const int hi   = (lane >> 4) & 1;    // half of wave (lanes 0-15 vs 16-31)
    const int l16  = lane & 15;
    const int rowTile = blockIdx.x * 32;
    const int colBase = blockIdx.y * 512 + wave * 64;

    const unsigned short* arow0 = Abf  + (size_t)(rowTile + l16) * KP;
    const unsigned short* arow1 = arow0 + (size_t)16 * KP;
    const unsigned short* brow0 = Btbf + (size_t)(colBase + l16) * KP;

    v8f acc0[4] = {};
    v8f acc1[4] = {};

    for (int k0 = 0; k0 < KP; k0 += 32) {
        // A 16x32 bf16 frags: lane(l16,hi) holds K runs [k0+8*hi, +8) and [k0+16+8*hi, +8)
        Frag a0, a1;
        a0.q[0] = *(const f4*)(arow0 + k0 + 8 * hi);
        a0.q[1] = *(const f4*)(arow0 + k0 + 16 + 8 * hi);
        a1.q[0] = *(const f4*)(arow1 + k0 + 8 * hi);
        a1.q[1] = *(const f4*)(arow1 + k0 + 16 + 8 * hi);
#pragma unroll
        for (int j = 0; j < 4; ++j) {
            // B 32x16 bf16 frag: lane(l16=n, hi) holds K runs [k0+8*hi, +8) and [k0+16+8*hi, +8)
            const unsigned short* brow = brow0 + (size_t)(j * 16) * KP;
            Frag b;
            b.q[0] = *(const f4*)(brow + k0 + 8 * hi);
            b.q[1] = *(const f4*)(brow + k0 + 16 + 8 * hi);
            acc0[j] = __builtin_amdgcn_wmma_f32_16x16x32_bf16(
                false, a0.v, false, b.v, (short)0, acc0[j], false, false);
            acc1[j] = __builtin_amdgcn_wmma_f32_16x16x32_bf16(
                false, a1.v, false, b.v, (short)0, acc1[j], false, false);
        }
    }

    // D layout: VGPR r, lane -> M = r + 8*hi, N = l16
#pragma unroll
    for (int j = 0; j < 4; ++j) {
        int col = colBase + j * 16 + l16;
        if (col < nstore) {
#pragma unroll
            for (int r = 0; r < 8; ++r) {
                int row = rowTile + r + 8 * hi;
                C[(size_t)row * ldc + col] = acc0[j][r];
                C[(size_t)(row + 16) * ldc + col] = acc1[j][r];
            }
        }
    }
}

// s_neigh[h, n] = X[n,:] . attn_neigh[h,:]
__global__ void k_scores_neigh(const float* __restrict__ X, const float* __restrict__ an,
                               float* __restrict__ s_neigh) {
    int idx = blockIdx.x * blockDim.x + threadIdx.x;
    if (idx >= H_ * NP) return;
    int h = idx / NP, n = idx % NP;
    float s = 0.f;
    if (n < N_) {
        const float* xr = X + (size_t)n * F_;
        const float* ar = an + h * F_;
        for (int f = 0; f < F_; ++f) s += xr[f] * ar[f];
    }
    s_neigh[idx] = s;
}

// s_self[h, bt] = feat[bt,:] . attn_self[h,:]
__global__ void k_scores_self(const float* __restrict__ feat, const float* __restrict__ as,
                              float* __restrict__ s_self) {
    int idx = blockIdx.x * blockDim.x + threadIdx.x;
    if (idx >= H_ * BT_) return;
    int h = idx / BT_, bt = idx % BT_;
    const float* fr = feat + (size_t)bt * F_;
    const float* ar = as + h * F_;
    float s = 0.f;
    for (int f = 0; f < F_; ++f) s += fr[f] * ar[f];
    s_self[idx] = s;
}

// Fused: masked-LeakyReLU scores -> softmax -> re-mask -> out = attn@X + bias, ReLU.
// One block per (b,t); heads sequential; scores staged in LDS.
__global__ __launch_bounds__(256)
void k_attn_out(const float* __restrict__ neigh, const float* __restrict__ s_self,
                const float* __restrict__ s_neigh, const float* __restrict__ X,
                const float* __restrict__ bias, float* __restrict__ out) {
    __shared__ float w[NP];      // 20032 B
    __shared__ float red[256];
    const int bt  = blockIdx.x;
    const int tid = threadIdx.x;
    const float* nrow = neigh + (size_t)bt * NP;

    for (int h = 0; h < H_; ++h) {
        const float ss = s_self[h * BT_ + bt];
        const float* snr = s_neigh + h * NP;

        // pass 1: d[n] = m*lrelu(ss+sn) + NEG_INF*(1-m); track max
        float lmax = -3.4e38f;
        for (int n = tid; n < N_; n += 256) {
            float v = ss + snr[n];
            v = (v > 0.f) ? v : ALPHA * v;
            float m = nrow[n];
            float d = m * v + NEG_INF_F * (1.0f - m);
            w[n] = d;
            lmax = fmaxf(lmax, d);
        }
        red[tid] = lmax; __syncthreads();
        for (int s = 128; s > 0; s >>= 1) {
            if (tid < s) red[tid] = fmaxf(red[tid], red[tid + s]);
            __syncthreads();
        }
        const float dmax = red[0];
        __syncthreads();

        // pass 2: w[n] = m * exp(d - dmax); accumulate sum of exp(d - dmax)
        float lsum = 0.f;
        for (int n = tid; n < N_; n += 256) {
            float p = expf(w[n] - dmax);
            lsum += p;
            w[n] = nrow[n] * p;     // re-mask: attn = neigh * softmax
        }
        red[tid] = lsum; __syncthreads();
        for (int s = 128; s > 0; s >>= 1) {
            if (tid < s) red[tid] += red[tid + s];
            __syncthreads();
        }
        const float inv = 1.0f / red[0];
        __syncthreads();

        // pass 3: out[f] = relu( (sum_n w[n]*X[n,f]) * inv + bias[h,f] )
        const int f = tid & 127, c = tid >> 7;
        float acc = 0.f;
        for (int n = c; n < N_; n += 2)
            acc += w[n] * X[(size_t)n * F_ + f];
        red[tid] = acc; __syncthreads();
        if (c == 0) {
            float tot = (red[f] + red[f + 128]) * inv + bias[h * F_ + f];
            out[(size_t)bt * (H_ * F_) + h * F_ + f] = fmaxf(tot, 0.f);
        }
        __syncthreads();
    }
}

extern "C" void kernel_launch(void* const* d_in, const int* in_sizes, int n_in,
                              void* d_out, int out_size, void* d_ws, size_t ws_size,
                              hipStream_t stream) {
    const float* node  = (const float*)d_in[0];   // [B,T,N]
    const float* X     = (const float*)d_in[1];   // [N,F]
    const float* A     = (const float*)d_in[2];   // [N,N]
    const float* aself = (const float*)d_in[3];   // [H,F]
    const float* aneig = (const float*)d_in[4];   // [H,F]
    const float* bias  = (const float*)d_in[5];   // [H,F]
    float* out = (float*)d_out;

    char* ws = (char*)d_ws;
    size_t off = 0;
    auto carve = [&](size_t bytes) { size_t o = off; off += (bytes + 255) & ~(size_t)255; return o; };

    unsigned short* nodeBF = (unsigned short*)(ws + carve((size_t)BT_    * KP * 2));
    unsigned short* AtBF   = (unsigned short*)(ws + carve((size_t)ATROWS * KP * 2));
    unsigned short* XtBF   = (unsigned short*)(ws + carve((size_t)XTROWS * KP * 2));
    float* feat    = (float*)(ws + carve((size_t)BT_ * F_ * 4));
    float* neigh   = (float*)(ws + carve((size_t)BT_ * NP * 4));
    float* s_neigh = (float*)(ws + carve((size_t)H_ * NP * 4));
    float* s_self  = (float*)(ws + carve((size_t)H_ * BT_ * 4));
    (void)ws_size; (void)in_sizes; (void)n_in; (void)out_size;

    // 1) precision/layout conversion
    {
        long long tot = (long long)BT_ * KP;
        k_cast_pad<<<(unsigned)((tot + 255) / 256), 256, 0, stream>>>(node, nodeBF, BT_, N_, KP);
    }
    {
        long long tot = (long long)ATROWS * KP;
        k_transpose_cast<<<(unsigned)((tot + 255) / 256), 256, 0, stream>>>(A, AtBF, ATROWS, N_, N_);
    }
    {
        long long tot = (long long)XTROWS * KP;
        k_transpose_cast<<<(unsigned)((tot + 255) / 256), 256, 0, stream>>>(X, XtBF, XTROWS, N_, F_);
    }

    // 2) WMMA GEMMs: feat = node@X ; neigh = node@A  (32-row M-tiles)
    k_gemm_bf16<<<dim3(BT_ / 32, 1),  256, 0, stream>>>(nodeBF, XtBF, feat,  F_, F_);
    k_gemm_bf16<<<dim3(BT_ / 32, 10), 256, 0, stream>>>(nodeBF, AtBF, neigh, NP, NP);

    // 3) attention scores
    k_scores_neigh<<<(H_ * NP + 255) / 256, 256, 0, stream>>>(X, aneig, s_neigh);
    k_scores_self <<<(H_ * BT_ + 255) / 256, 256, 0, stream>>>(feat, aself, s_self);

    // 4) fused softmax + output
    k_attn_out<<<BT_, 256, 0, stream>>>(neigh, s_self, s_neigh, X, bias, out);
}
